// SharedExpertMoE_49675591745705
// MI455X (gfx1250) — matrix operand order repaired and checked by reading, not compiled, not run on hardware
//
#include <hip/hip_runtime.h>
#include <hip/hip_bf16.h>

// ---------------- types ----------------
typedef __attribute__((ext_vector_type(16))) __bf16       bf16x16;
typedef __attribute__((ext_vector_type(8)))  float        floatx8;
typedef __attribute__((ext_vector_type(4)))  float        floatx4;
typedef __attribute__((ext_vector_type(4)))  unsigned int uintx4;
typedef __attribute__((ext_vector_type(2)))  unsigned int uintx2;

union FragAB { uintx4 q[2]; bf16x16 v; };
union AccU   { floatx8 v; float f[8]; };
union F4U    { floatx4 v; float f[4]; unsigned int u[4]; };

// Pack two fp32 -> two bf16 (truncate) in ONE v_perm_b32.
// result = { hi[31:16], lo[31:16] }
static __device__ __forceinline__ unsigned int pk_bf16(float lo, float hi) {
  return __builtin_amdgcn_perm(__builtin_bit_cast(unsigned int, hi),
                               __builtin_bit_cast(unsigned int, lo),
                               0x07060302u);
}
static __device__ __forceinline__ unsigned short f2bf(float f) {
  return (unsigned short)(__builtin_bit_cast(unsigned int, f) >> 16);
}
// XOR swizzle key for 16B k-chunks keyed on row/col: spreads fragment
// ds_load_b128 across all 64 LDS banks (removes 4-way read conflicts).
static __device__ __forceinline__ int swzkey(int rc) {
  return ((rc >> 2) ^ (rc >> 4)) & 3;
}

// ---------------- problem constants ----------------
#define NTOK 2048   // B*L
#define DIM  1024
#define NEXP 8
#define H1D  4096
#define H2D  2048

// =====================================================================
// x (fp32) -> bf16
// =====================================================================
__global__ __launch_bounds__(256) void cvt_x_kernel(const float* __restrict__ x,
                                                    unsigned short* __restrict__ xb,
                                                    int n4) {
  int i = blockIdx.x * 256 + threadIdx.x;
  if (i < n4) {
    floatx4 f = ((const floatx4*)x)[i];
    uintx2 p;
    p.x = pk_bf16(f.x, f.y);
    p.y = pk_bf16(f.z, f.w);
    ((uintx2*)xb)[i] = p;
  }
}

// =====================================================================
// Gating: logits = x @ Wg, top-2 softmax -> cw[tok][8]; top-1 counts.
// One wave (32 lanes) per token.
// =====================================================================
__global__ __launch_bounds__(128) void moe_gate_kernel(const float* __restrict__ x,
                                                       const float* __restrict__ Wg,
                                                       float* __restrict__ cw,
                                                       unsigned int* __restrict__ counts) {
  const int tok  = blockIdx.x * 4 + (threadIdx.x >> 5);
  const int lane = threadIdx.x & 31;
  float acc[8] = {0.f, 0.f, 0.f, 0.f, 0.f, 0.f, 0.f, 0.f};
  const float* xr = x + (size_t)tok * DIM;
  for (int d = lane; d < DIM; d += 32) {
    float xv = xr[d];
    floatx4 w0 = *(const floatx4*)(Wg + (size_t)d * NEXP);
    floatx4 w1 = *(const floatx4*)(Wg + (size_t)d * NEXP + 4);
    acc[0] += xv * w0.x; acc[1] += xv * w0.y; acc[2] += xv * w0.z; acc[3] += xv * w0.w;
    acc[4] += xv * w1.x; acc[5] += xv * w1.y; acc[6] += xv * w1.z; acc[7] += xv * w1.w;
  }
#pragma unroll
  for (int off = 16; off > 0; off >>= 1)
#pragma unroll
    for (int e = 0; e < 8; e++) acc[e] += __shfl_down(acc[e], off, 32);
  if (lane == 0) {
    int i0 = 0;
#pragma unroll
    for (int e = 1; e < 8; e++) if (acc[e] > acc[i0]) i0 = e;
    int i1 = (i0 == 0) ? 1 : 0;
#pragma unroll
    for (int e = 0; e < 8; e++) if (e != i0 && acc[e] > acc[i1]) i1 = e;
    float z  = __expf(acc[i1] - acc[i0]);          // <= 1, stable
    float g0 = 1.f / (1.f + z);
    float g1 = z / (1.f + z);
    float row[8] = {0.f, 0.f, 0.f, 0.f, 0.f, 0.f, 0.f, 0.f};
    row[i0] += g0; row[i1] += g1;
    floatx4* dst = (floatx4*)(cw + (size_t)tok * NEXP);
    floatx4 r0; r0.x = row[0]; r0.y = row[1]; r0.z = row[2]; r0.w = row[3];
    floatx4 r1; r1.x = row[4]; r1.y = row[5]; r1.z = row[6]; r1.w = row[7];
    dst[0] = r0; dst[1] = r1;
    atomicAdd(counts + i0, 1u);
  }
}

__global__ void lb_loss_kernel(const unsigned int* __restrict__ counts,
                               float* __restrict__ loss_out) {
  if (threadIdx.x == 0 && blockIdx.x == 0) {
    float tot = 0.f, c[8];
#pragma unroll
    for (int e = 0; e < 8; e++) { c[e] = (float)counts[e]; tot += c[e]; }
    float loss = 0.f;
#pragma unroll
    for (int e = 0; e < 8; e++) { float f = c[e] / tot - 0.125f; loss += f * f; }
    loss_out[0] = loss * 0.125f;
  }
}

// =====================================================================
// Tiled bf16 WMMA GEMM.
//   block = 256 threads (8 waves, 2 x 4), block tile 128x128, BK=32
//   wave tile 64x32 -> 4x2 v_wmma_f32_16x16x32_bf16 per K-step (x2 halves
//   for the fused-SwiGLU variant).
//   fp32 weights -> bf16 during global->LDS stage via packed v_perm_b32;
//   B stored K-contiguous per column; XOR-swizzled 16B k-chunks make all
//   fragment ds_load_b128 bank-conflict-free.
//   Software double buffering: glb->regs | WMMA | regs->LDS | barrier.
// EPI: 0 = dual-half accum + fused SwiGLU -> bf16 H (Ncol = half width)
//      1 = plain bf16 store
//      2 = out[row][col] += gate * acc  (gate = cw[row][expert] or cscale)
// =====================================================================
template <int EPI>
__global__ __launch_bounds__(256) void gemm_bf16_wmma(
    const unsigned short* __restrict__ A,   // bf16 [M][K], row-major
    const float* __restrict__ Bw,           // fp32 [K][ldb], row-major
    void* __restrict__ Cout,
    int K, int Ncol, int ldb,
    const float* __restrict__ gate, int expert, float cscale) {
  constexpr int NB = (EPI == 0) ? 2 : 1;
  __shared__ unsigned short As[2][128][32];
  __shared__ unsigned short Bs[2][NB][128][32];

  const int t    = threadIdx.x;
  const int row0 = blockIdx.y * 128;
  const int col0 = blockIdx.x * 128;
  const int lane = t & 31;
  const int w    = t >> 5;
  const int wr   = (w >> 2) * 64;   // wave row: 0 / 64
  const int wc   = (w & 3) * 32;    // wave col: 0 / 32 / 64 / 96
  const int hi   = lane >> 4;       // 0: lanes 0-15, 1: lanes 16-31
  const int ll   = lane & 15;

  floatx8 acc[NB][4][2];
  floatx8 zero = {0.f, 0.f, 0.f, 0.f, 0.f, 0.f, 0.f, 0.f};
#pragma unroll
  for (int h = 0; h < NB; h++)
#pragma unroll
    for (int mi = 0; mi < 4; mi++)
#pragma unroll
      for (int ni = 0; ni < 2; ni++) acc[h][mi][ni] = zero;

  // staging registers
  uintx4 rA[2];
  F4U    rB[NB][4];

  // B staging micro-tile: 4 k-rows x 4 cols per thread
  const int kq = (t >> 5) * 4;      // 0..28
  const int cq = (t & 31) * 4;      // 0..124
  const int KT = K >> 5;

  auto g_load = [&](int kt) {
    const int k0 = kt << 5;
#pragma unroll
    for (int i = 0; i < 2; i++) {
      int q = t + (i << 8);                  // 0..511 uint4 chunks of A tile
      int r = q >> 2;
      int c = (q & 3) << 3;
      rA[i] = *(const uintx4*)(A + (size_t)(row0 + r) * K + k0 + c);
    }
#pragma unroll
    for (int h = 0; h < NB; h++) {
      const float* src = Bw + (size_t)h * Ncol;   // g-half lives Ncol columns later
#pragma unroll
      for (int kk = 0; kk < 4; kk++)
        rB[h][kk].v = *(const floatx4*)(src + (size_t)(k0 + kq + kk) * ldb + col0 + cq);
      if (k0 + 32 < K)   // L2 prefetch of the K-tile after next
        __builtin_prefetch(src + (size_t)(k0 + 32 + kq) * ldb + col0 + cq, 0, 1);
    }
  };

  auto s_store = [&](int buf) {
#pragma unroll
    for (int i = 0; i < 2; i++) {
      int q  = t + (i << 8);
      int r  = q >> 2;
      int ck = q & 3;                              // logical 16B k-chunk
      *(uintx4*)&As[buf][r][((ck ^ swzkey(r)) << 3)] = rA[i];
    }
#pragma unroll
    for (int h = 0; h < NB; h++)
#pragma unroll
      for (int cc = 0; cc < 4; cc++) {
        const int col = cq + cc;
        const int kk  = (((kq >> 3) ^ swzkey(col)) << 3) | (kq & 4);
        uintx2 p;
        p.x = pk_bf16(rB[h][0].f[cc], rB[h][1].f[cc]);   // k, k+1
        p.y = pk_bf16(rB[h][2].f[cc], rB[h][3].f[cc]);   // k+2, k+3
        *(uintx2*)&Bs[buf][h][col][kk] = p;
      }
  };

  auto compute = [&](int buf) {
    FragAB fa[4];
#pragma unroll
    for (int mi = 0; mi < 4; mi++) {
      int r = wr + mi * 16 + ll;
      int k = swzkey(r);
      // lane<16: V0-3=K0..7, V4-7=K16..23 ; lane>=16: K8..15 / K24..31
      fa[mi].q[0] = *(const uintx4*)&As[buf][r][((hi ^ k) << 3)];
      fa[mi].q[1] = *(const uintx4*)&As[buf][r][(((2 + hi) ^ k) << 3)];
    }
#pragma unroll
    for (int h = 0; h < NB; h++) {
      FragAB fb[2];
#pragma unroll
      for (int ni = 0; ni < 2; ni++) {
        int c = wc + ni * 16 + ll;
        int k = swzkey(c);
        // lane<16: V0-3=K0..7, V4-7=K16..23 ; lane>=16: K8..15 / K24..31
        fb[ni].q[0] = *(const uintx4*)&Bs[buf][h][c][((hi ^ k) << 3)];
        fb[ni].q[1] = *(const uintx4*)&Bs[buf][h][c][(((2 + hi) ^ k) << 3)];
      }
#pragma unroll
      for (int mi = 0; mi < 4; mi++)
#pragma unroll
        for (int ni = 0; ni < 2; ni++)
          acc[h][mi][ni] = __builtin_amdgcn_wmma_f32_16x16x32_bf16(
              false, fa[mi].v, false, fb[ni].v, (short)0, acc[h][mi][ni],
              false, false);
    }
  };

  g_load(0);
  s_store(0);
  __syncthreads();
  for (int kt = 0; kt < KT; ++kt) {
    if (kt + 1 < KT) g_load(kt + 1);   // VMEM overlaps the WMMA stream
    compute(kt & 1);
    if (kt + 1 < KT) s_store((kt + 1) & 1);
    __syncthreads();
  }

  // ---------------- epilogue ----------------
#pragma unroll
  for (int mi = 0; mi < 4; mi++) {
#pragma unroll
    for (int ni = 0; ni < 2; ni++) {
      const int gr = row0 + wr + mi * 16 + hi * 8;  // C layout: V_j = row gr+j
      const int gc = col0 + wc + ni * 16 + ll;
      if constexpr (EPI == 0) {
        AccU ua, ug;
        ua.v = acc[0][mi][ni];
        ug.v = acc[1][mi][ni];
        unsigned short* H = (unsigned short*)Cout;
#pragma unroll
        for (int j = 0; j < 8; j++) {
          float a = ua.f[j], g = ug.f[j];
          float s = a / (1.f + __expf(-a));         // silu
          H[(size_t)(gr + j) * Ncol + gc] = f2bf(s * g);
        }
      } else if constexpr (EPI == 1) {
        AccU u; u.v = acc[0][mi][ni];
        unsigned short* T = (unsigned short*)Cout;
#pragma unroll
        for (int j = 0; j < 8; j++)
          T[(size_t)(gr + j) * Ncol + gc] = f2bf(u.f[j]);
      } else {
        AccU u; u.v = acc[0][mi][ni];
        float* O = (float*)Cout;
#pragma unroll
        for (int j = 0; j < 8; j++) {
          float gv = gate ? gate[(size_t)(gr + j) * NEXP + expert] : cscale;
          size_t idx = (size_t)(gr + j) * Ncol + gc;
          O[idx] += gv * u.f[j];
        }
      }
    }
  }
}

// =====================================================================
// host launch
// =====================================================================
extern "C" void kernel_launch(void* const* d_in, const int* in_sizes, int n_in,
                              void* d_out, int out_size, void* d_ws, size_t ws_size,
                              hipStream_t stream) {
  (void)in_sizes; (void)n_in; (void)ws_size;
  const float* x   = (const float*)d_in[0];
  const float* Wg  = (const float*)d_in[1];
  const float* We1 = (const float*)d_in[2];
  const float* We2 = (const float*)d_in[3];
  const float* We3 = (const float*)d_in[4];
  const float* Ws1 = (const float*)d_in[5];
  const float* Ws2 = (const float*)d_in[6];
  float* out = (float*)d_out;

  char* wsb = (char*)d_ws;
  unsigned short* xb   = (unsigned short*)(wsb);                        // 4 MB
  unsigned short* Hbuf = (unsigned short*)(wsb + (4u  << 20));          // 16 MB
  unsigned short* Tbuf = (unsigned short*)(wsb + (20u << 20));          // 4 MB
  float*          cw   = (float*)(wsb + (24u << 20));                   // 64 KB
  unsigned int* counts = (unsigned int*)(wsb + (24u << 20) + NTOK * NEXP * sizeof(float));

  hipMemsetAsync(d_out, 0, (size_t)out_size * sizeof(float), stream);
  hipMemsetAsync(counts, 0, NEXP * sizeof(unsigned int), stream);

  cvt_x_kernel<<<(NTOK * DIM / 4 + 255) / 256, 256, 0, stream>>>(x, xb, NTOK * DIM / 4);
  moe_gate_kernel<<<NTOK / 4, 128, 0, stream>>>(x, Wg, cw, counts);
  lb_loss_kernel<<<1, 32, 0, stream>>>(counts, out + (size_t)NTOK * DIM);

  const dim3 blk(256);
  // ----- shared experts: out += 0.5 * (swiglu(x@Ws1[s]) @ Ws2[s]) -----
  for (int s = 0; s < 2; s++) {
    gemm_bf16_wmma<0><<<dim3(H2D / 128, NTOK / 128), blk, 0, stream>>>(
        xb, Ws1 + (size_t)s * DIM * 2 * H2D, Hbuf,
        DIM, H2D, 2 * H2D, nullptr, 0, 0.f);
    gemm_bf16_wmma<2><<<dim3(DIM / 128, NTOK / 128), blk, 0, stream>>>(
        Hbuf, Ws2 + (size_t)s * H2D * DIM, out,
        H2D, DIM, DIM, nullptr, 0, 0.5f);
  }
  // ----- routed experts: out += cw[:,e] * ((swiglu(x@We1[e])@We2[e])@We3[e]) -----
  for (int e = 0; e < NEXP; e++) {
    gemm_bf16_wmma<0><<<dim3(H1D / 128, NTOK / 128), blk, 0, stream>>>(
        xb, We1 + (size_t)e * DIM * 2 * H1D, Hbuf,
        DIM, H1D, 2 * H1D, nullptr, 0, 0.f);
    gemm_bf16_wmma<1><<<dim3(DIM / 128, NTOK / 128), blk, 0, stream>>>(
        Hbuf, We2 + (size_t)e * H1D * DIM, Tbuf,
        H1D, DIM, DIM, nullptr, 0, 0.f);
    gemm_bf16_wmma<2><<<dim3(DIM / 128, NTOK / 128), blk, 0, stream>>>(
        Tbuf, We3 + (size_t)e * DIM * DIM, out,
        DIM, DIM, DIM, cw, e, 0.f);
  }
}
